// _MahalanobisEnsembleLoss_71588514889740
// MI455X (gfx1250) — matrix operand order repaired and checked by reading, not compiled, not run on hardware
//
#include <hip/hip_runtime.h>
#include <stdint.h>

typedef __attribute__((ext_vector_type(16))) _Float16 v16h;
typedef __attribute__((ext_vector_type(8)))  float    v8f;

#define L_DIM 4
#define N_DIM 16384
#define D_DIM 512
#define C_DIM 1000

#define TILE    128                      // macro-tile edge of Gram matrix G
#define NBLK    (D_DIM / TILE)           // 4 block-rows of G
#define NTILES  (NBLK * (NBLK + 1) / 2)  // 10 upper-triangular macro-tiles
#define NSPLIT  16                       // split of N (K dimension) across WGs
#define SLICE   (N_DIM / NSPLIT)         // 1024 samples per WG
#define KC      32                       // n-chunk per iteration (WMMA K)
#define CHUNKS  (SLICE / KC)             // 32
#define PITCH   34                       // halfs per LDS row = 68B (odd-DWORD -> ~2-way write conflicts)

// ---- WMMA operand fragment builders (per CDNA5 ISA 16-bit layouts) ----
// A (16x32, M=lane%16 row, K packed): lanes 0-15 hold K {0..7,16..23}, lanes 16-31 hold K {8..15,24..31}
__device__ __forceinline__ v16h frag_A(const _Float16* Z, int rowBase, int lane) {
  const int half = lane >> 4, m = lane & 15;
  const uint32_t* r = (const uint32_t*)(Z + (size_t)(rowBase + m) * PITCH);
  const int o0 = half * 4;       // halfs {0..7} or {8..15}
  const int o1 = 8 + half * 4;   // halfs {16..23} or {24..31}
  union { v16h v; uint32_t u[8]; } f;
#pragma unroll
  for (int p = 0; p < 4; ++p) { f.u[p] = r[o0 + p]; f.u[4 + p] = r[o1 + p]; }
  return f.v;
}
// B (32x16, N=lane%16 col): lanes 0-15 hold K {0..7,16..23}, lanes 16-31 hold K {8..15,24..31} (K-pairs packed per VGPR)
__device__ __forceinline__ v16h frag_B(const _Float16* Z, int colBase, int lane) {
  const int half = lane >> 4, nn = lane & 15;
  const uint32_t* r = (const uint32_t*)(Z + (size_t)(colBase + nn) * PITCH);
  const int o = half * 4;
  union { v16h v; uint32_t u[8]; } f;
#pragma unroll
  for (int p = 0; p < 4; ++p) { f.u[p] = r[o + p]; f.u[4 + p] = r[8 + o + p]; }
  return f.v;
}

__global__ __launch_bounds__(256)
void mahal_gram_kernel(const float* __restrict__ feature,
                       const float* __restrict__ mean,
                       const float* __restrict__ icov,
                       const float* __restrict__ weight,
                       const long long* __restrict__ idx,
                       float* __restrict__ partials) {
  __shared__ _Float16 Zhi[2 * TILE * PITCH];   // hi f16 of z, transposed [row(d)][n]
  __shared__ _Float16 Zlo[2 * TILE * PITCH];   // lo f16 residual of z
  __shared__ float    red[256];

  const int tid  = threadIdx.x;
  const int lane = tid & 31;
  const int w    = tid >> 5;               // wave id 0..7

  int bx = blockIdx.x;
  const int slice = bx % NSPLIT; bx /= NSPLIT;
  int tile = bx % NTILES; const int l = bx / NTILES;

  // triangular tile id -> (i, j), i <= j over NBLK x NBLK blocks
  int i = 0; { int t = tile; while (t >= NBLK - i) { t -= NBLK - i; ++i; } tile = t; }
  const int j     = i + tile;
  const int iBase = i * TILE, jBase = j * TILE;
  const bool diag = (i == j);
  const int rowsTotal = diag ? TILE : 2 * TILE;   // LDS rows: d-block (+ e-block if off-diag)
  const int shQ   = diag ? 5 : 6;                 // log2(rowsTotal/4)
  const int rowsQ = rowsTotal >> 2;
  const int tasks = KC * rowsQ;                   // one float4 of d per task
  const int Boff  = diag ? 0 : TILE;              // B operand row offset in LDS

  const float* featL = feature + (size_t)l * N_DIM * D_DIM;
  const float* meanL = mean    + (size_t)l * C_DIM * D_DIM;
  const float* AinvL = icov    + (size_t)l * D_DIM * D_DIM;
  const int n00 = slice * SLICE;

  // wave -> 2 (rows) x 4 (cols) subtiles of the 8x8 16x16-subtile grid
  const int wr = (w >> 1) * 2;   // subtile row base: 0,2,4,6
  const int wc = (w & 1) * 4;    // subtile col base: 0 or 4

  v8f acc[2][4] = {};            // f32 accumulators (WMMA C/D)

  for (int ck = 0; ck < CHUNKS; ++ck) {
    const int n0 = n00 + ck * KC;

    // ---- producer: z = feature - mean[idx], split into f16 hi/lo, transpose into LDS ----
    for (int t = tid; t < tasks; t += 256) {
      const int nl  = t >> shQ;                 // local n 0..31
      const int dg  = t & (rowsQ - 1);          // float4 group (consecutive tids -> coalesced d)
      const int rBlk = dg * 4;                  // local LDS row of first element
      const int dG = (rBlk < TILE) ? (iBase + rBlk) : (jBase + rBlk - TILE);
      const int n  = n0 + nl;
      const int g  = (int)idx[n];
      const float* fp = featL + (size_t)n * D_DIM + dG;
      const float4 f4 = *(const float4*)fp;
      const float4 m4 = *(const float4*)(meanL + (size_t)g * D_DIM + dG);
      if (ck + 1 < CHUNKS)
        __builtin_prefetch(fp + (size_t)KC * D_DIM, 0, 1);   // next chunk feature stream
      float z[4] = { f4.x - m4.x, f4.y - m4.y, f4.z - m4.z, f4.w - m4.w };
#pragma unroll
      for (int c = 0; c < 4; ++c) {
        const _Float16 h  = (_Float16)z[c];
        const _Float16 lo = (_Float16)(z[c] - (float)h);
        Zhi[(size_t)(rBlk + c) * PITCH + nl] = h;
        Zlo[(size_t)(rBlk + c) * PITCH + nl] = lo;
      }
    }
    __syncthreads();

    // ---- consumer: G += Hi^T Hi + Hi^T Lo + Lo^T Hi via v_wmma_f32_16x16x32_f16 ----
    v16h ahi[2], alo[2];
#pragma unroll
    for (int rr = 0; rr < 2; ++rr) {
      ahi[rr] = frag_A(Zhi, (wr + rr) * 16, lane);
      alo[rr] = frag_A(Zlo, (wr + rr) * 16, lane);
    }
#pragma unroll
    for (int cc = 0; cc < 4; ++cc) {
      const int cb = Boff + (wc + cc) * 16;
      const v16h bhi = frag_B(Zhi, cb, lane);
      const v16h blo = frag_B(Zlo, cb, lane);
#pragma unroll
      for (int rr = 0; rr < 2; ++rr) {
        acc[rr][cc] = __builtin_amdgcn_wmma_f32_16x16x32_f16(
            false, ahi[rr], false, bhi, (short)0, acc[rr][cc], false, false);
        acc[rr][cc] = __builtin_amdgcn_wmma_f32_16x16x32_f16(
            false, ahi[rr], false, blo, (short)0, acc[rr][cc], false, false);
        acc[rr][cc] = __builtin_amdgcn_wmma_f32_16x16x32_f16(
            false, alo[rr], false, bhi, (short)0, acc[rr][cc], false, false);
      }
    }
    __syncthreads();
  }

  // ---- epilogue: s = sum over tile of A[d,e]*G[d,e] (+ mirror for off-diagonal tiles) ----
  // C/D layout: VGPR v, lanes 0-15: M=v, N=lane; lanes 16-31: M=v+8, N=lane-16
  const int half = lane >> 4, nn = lane & 15;
  float s = 0.f;
#pragma unroll
  for (int rr = 0; rr < 2; ++rr) {
#pragma unroll
    for (int cc = 0; cc < 4; ++cc) {
      const int dB = iBase + (wr + rr) * 16 + half * 8;
      const int eG = jBase + (wc + cc) * 16 + nn;
#pragma unroll
      for (int v = 0; v < 8; ++v) {
        const int dG = dB + v;
        float aw = AinvL[(size_t)dG * D_DIM + eG];
        if (!diag) aw += AinvL[(size_t)eG * D_DIM + dG];
        s += acc[rr][cc][v] * aw;
      }
    }
  }

  red[tid] = s;
  __syncthreads();
#pragma unroll
  for (int off = 128; off > 0; off >>= 1) {
    if (tid < off) red[tid] += red[tid + off];
    __syncthreads();
  }
  if (tid == 0)
    partials[blockIdx.x] = red[0] * (weight[l] / (float)N_DIM);
}

__global__ __launch_bounds__(256)
void mahal_reduce_kernel(const float* __restrict__ partials, int n, float* __restrict__ out) {
  __shared__ float red[256];
  float s = 0.f;
  for (int k = threadIdx.x; k < n; k += 256) s += partials[k];
  red[threadIdx.x] = s;
  __syncthreads();
#pragma unroll
  for (int off = 128; off > 0; off >>= 1) {
    if (threadIdx.x < off) red[threadIdx.x] += red[threadIdx.x + off];
    __syncthreads();
  }
  if (threadIdx.x == 0) out[0] = red[0];
}

extern "C" void kernel_launch(void* const* d_in, const int* in_sizes, int n_in,
                              void* d_out, int out_size, void* d_ws, size_t ws_size,
                              hipStream_t stream) {
  const float*     feature = (const float*)d_in[0];
  const float*     mean    = (const float*)d_in[1];
  const float*     icov    = (const float*)d_in[2];
  const float*     weight  = (const float*)d_in[3];
  const long long* idx     = (const long long*)d_in[4];
  float* partials = (float*)d_ws;

  const int NB = L_DIM * NTILES * NSPLIT;   // 640 workgroups
  mahal_gram_kernel<<<NB, 256, 0, stream>>>(feature, mean, icov, weight, idx, partials);
  mahal_reduce_kernel<<<1, 256, 0, stream>>>(partials, NB, (float*)d_out);

  (void)in_sizes; (void)n_in; (void)out_size; (void)ws_size;
}
